// CausalSelfAttention_28028956574316
// MI455X (gfx1250) — matrix operand order repaired and checked by reading, not compiled, not run on hardware
//
#include <hip/hip_runtime.h>
#include <hip/hip_bf16.h>
#include <math.h>

typedef __attribute__((ext_vector_type(16))) _Float16 v16h;
typedef __attribute__((ext_vector_type(8)))  _Float16 v8h;
typedef __attribute__((ext_vector_type(8)))  float    v8f;

#define BB  4
#define TT  2048
#define CC  768
#define NHH 12
#define HDD 64
#define MM  (BB*TT)   // 8192
#define N3  (3*CC)    // 2304

#define LDSPITCH 40   // 32 f16 + 8 pad: 80B row stride, 16B aligned, bank-clean

static __device__ inline v16h combine8(v8h lo, v8h hi) {
  v16h r;
#pragma unroll
  for (int i = 0; i < 8; ++i) { r[i] = lo[i]; r[i + 8] = hi[i]; }
  return r;
}

static __device__ inline v8f wmma_f16(v16h a, v16h b, v8f c) {
  return __builtin_amdgcn_wmma_f32_16x16x32_f16(false, a, false, b, (short)0, c,
                                                false, false);
}

// A-fragment load (contiguous k order): row = m, two 8-halfword chunks at
// K = h*16 and h*16 + 8 within the 32-wide k-chunk.
static __device__ inline v16h load_afrag(const _Float16* rowp, int k0, int half) {
  v8h lo = *(const v8h*)(rowp + k0 + half * 16);
  v8h hi = *(const v8h*)(rowp + k0 + half * 16 + 8);
  return combine8(lo, hi);
}

// Per-lane async global->LDS copy of 16 bytes (ASYNCcnt-tracked).
static __device__ inline void async_copy_b128(const _Float16* gp, unsigned lds_off) {
  asm volatile("global_load_async_to_lds_b128 %0, %1, off"
               :: "v"(lds_off), "v"(gp) : "memory");
}

static __device__ inline void wait_async0() {
  asm volatile("s_wait_asynccnt 0" ::: "memory");
}

// ---------------- conversion / transpose helpers ----------------

__global__ void k_f32_to_f16(const float* __restrict__ in,
                             _Float16* __restrict__ out, int n) {
  int i = blockIdx.x * blockDim.x + threadIdx.x;
  if (i < n) out[i] = (_Float16)in[i];
}

// in: K x N (row-major), out: N x K (row-major) f16
__global__ void k_transpose_f16(const float* __restrict__ in,
                                _Float16* __restrict__ out, int K, int N) {
  int i = blockIdx.x * blockDim.x + threadIdx.x;
  if (i >= K * N) return;
  int k = i / N, n = i - k * N;
  out[(size_t)n * K + k] = (_Float16)in[i];
}

// ---------------- QKV projection GEMM ----------------
// xh: (M=8192, K=768) f16 ; Wt: (N=2304, K=768) f16 (transposed W_attn)
// Block: 128M x 128N, 8 waves (4 M-pos x 2 N-pos), wave tile 32M x 64N.
// B tile (128N x 32K) staged in LDS via async global->LDS, double-buffered.
// Writes Q,K -> (B,NH,T,HD) f16 and V -> transposed (B,NH,HD,T) f16.
__global__ __launch_bounds__(256) void k_qkv_gemm(
    const _Float16* __restrict__ xh, const _Float16* __restrict__ Wt,
    const float* __restrict__ bias, _Float16* __restrict__ Qd,
    _Float16* __restrict__ Kd, _Float16* __restrict__ Vtd) {
  __shared__ __align__(64) _Float16 ldsB[2][128][LDSPITCH];
  const int tid = threadIdx.x;
  const int l = tid & 31;
  const int wv = tid >> 5;
  const int ln = l & 15, half = l >> 4;
  const int m0 = blockIdx.x * 128 + (wv & 3) * 32;
  const int n0b = blockIdx.y * 128;
  const int nf = (wv >> 2) * 64;   // wave's N base within block tile

  const _Float16* Ap0 = xh + (size_t)(m0 + ln) * CC;
  const _Float16* Ap1 = xh + (size_t)(m0 + 16 + ln) * CC;
  const v8f zero = {0.f, 0.f, 0.f, 0.f, 0.f, 0.f, 0.f, 0.f};
  v8f acc[2][4] = {{zero, zero, zero, zero}, {zero, zero, zero, zero}};

  // stage 128 rows x 32 f16 (8KB): 512 b128 segments, 2 per thread
  auto stage = [&](int buf, int k0) {
#pragma unroll
    for (int s = 0; s < 2; ++s) {
      int i = tid + 256 * s;
      int nrow = i >> 2, seg = i & 3;
      const _Float16* gp = Wt + (size_t)(n0b + nrow) * CC + k0 + seg * 8;
      unsigned la = (unsigned)(size_t)&ldsB[buf][nrow][seg * 8];
      async_copy_b128(gp, la);
    }
  };

  stage(0, 0);
  for (int k0 = 0; k0 < CC; k0 += 32) {
    int buf = (k0 >> 5) & 1;
    wait_async0();
    __syncthreads();                     // all waves' async copies landed
    if (k0 + 32 < CC) stage(buf ^ 1, k0 + 32);

    v16h a0 = load_afrag(Ap0, k0, half);
    v16h a1 = load_afrag(Ap1, k0, half);
#pragma unroll
    for (int j = 0; j < 4; ++j) {
      v16h b = *(const v16h*)&ldsB[buf][nf + j * 16 + ln][half * 16];
      acc[0][j] = wmma_f16(a0, b, acc[0][j]);
      acc[1][j] = wmma_f16(a1, b, acc[1][j]);
    }
    __syncthreads();                     // reads drained before buf reuse
  }

#pragma unroll
  for (int j = 0; j < 4; ++j) {
    int n = n0b + nf + j * 16 + ln;
    float bv = bias[n];
    int sec = n / CC;
    int cin = n - sec * CC;
    int head = cin >> 6;
    int d = cin & 63;
#pragma unroll
    for (int i = 0; i < 2; ++i) {
#pragma unroll
      for (int r = 0; r < 8; ++r) {
        int row = m0 + i * 16 + r + 8 * half;
        int b_ = row >> 11;       // / T
        int t = row & (TT - 1);
        _Float16 hv = (_Float16)(acc[i][j][r] + bv);
        size_t bh = (size_t)b_ * NHH + head;
        if (sec == 0)
          Qd[(bh * TT + t) * HDD + d] = hv;
        else if (sec == 1)
          Kd[(bh * TT + t) * HDD + d] = hv;
        else
          Vtd[(bh * HDD + d) * TT + t] = hv;
      }
    }
  }
}

// ---------------- flash attention (one wave per 32-query tile) ----------------
__global__ __launch_bounds__(128) void k_attn(const _Float16* __restrict__ Qd,
                                              const _Float16* __restrict__ Kd,
                                              const _Float16* __restrict__ Vtd,
                                              _Float16* __restrict__ Yh) {
  // per wave: 2 query sub-tiles x (16 x 32) P staging
  __shared__ __align__(32) _Float16 lds[4][2][16 * 32];
  const int l = threadIdx.x & 31;
  const int wv = threadIdx.x >> 5;
  const int ln = l & 15, half = l >> 4;
  int task = blockIdx.x * 4 + wv;
  int bh = task >> 6;            // / (T/32)
  int q0 = (task & 63) * 32;     // multiple of 32
  int b_ = bh / NHH, head = bh - b_ * NHH;

  const _Float16* Qp = Qd + (size_t)bh * TT * HDD;
  const _Float16* Kp = Kd + (size_t)bh * TT * HDD;
  const _Float16* Vp = Vtd + (size_t)bh * HDD * TT;

  // Q A-fragments: 2 query tiles x 2 k-chunks of the 64-wide head dim
  v16h aq[2][2];
#pragma unroll
  for (int i = 0; i < 2; ++i) {
    const _Float16* qr = Qp + (size_t)(q0 + i * 16 + ln) * HDD;
#pragma unroll
    for (int c = 0; c < 2; ++c) aq[i][c] = load_afrag(qr, c * 32, half);
  }

  const v8f zero = {0.f, 0.f, 0.f, 0.f, 0.f, 0.f, 0.f, 0.f};
  v8f acc[2][4] = {{zero, zero, zero, zero}, {zero, zero, zero, zero}};
  float m_i[2][8], l_i[2][8];
#pragma unroll
  for (int i = 0; i < 2; ++i)
#pragma unroll
    for (int r = 0; r < 8; ++r) { m_i[i][r] = -INFINITY; l_i[i][r] = 0.f; }

  for (int kb = 0; kb < q0 + 32; kb += 32) {
    // K B-fragments, shared by both query tiles: 2 key sub-tiles x 2 chunks
    v16h kf[2][2];
#pragma unroll
    for (int st = 0; st < 2; ++st)
#pragma unroll
      for (int c = 0; c < 2; ++c)
        kf[st][c] = *(const v16h*)(Kp + (size_t)(kb + st * 16 + ln) * HDD +
                                   c * 32 + half * 16);

    // S for both tiles (skip fully-masked diagonal sub-tiles)
    v8f s[2][2];
#pragma unroll
    for (int i = 0; i < 2; ++i) {
      int qi = q0 + i * 16;
      s[i][0] = zero;
#pragma unroll
      for (int c = 0; c < 2; ++c) s[i][0] = wmma_f16(aq[i][c], kf[0][c], s[i][0]);
      bool act1 = kb < qi;  // sub-tile 1 has any unmasked column
      s[i][1] = zero;
      if (act1) {
#pragma unroll
        for (int c = 0; c < 2; ++c)
          s[i][1] = wmma_f16(aq[i][c], kf[1][c], s[i][1]);
      }
    }

    // V B-fragments, shared by both query tiles
    v16h vf[4];
#pragma unroll
    for (int j = 0; j < 4; ++j)
      vf[j] = *(const v16h*)(Vp + (size_t)(j * 16 + ln) * TT + kb + half * 16);

    // per query tile: mask + online softmax + P relayout + PV wmma
#pragma unroll
    for (int i = 0; i < 2; ++i) {
      int qi = q0 + i * 16;
      bool act1 = kb < qi;  // when false, sub-tile 1 is entirely above diagonal
      float p0[8], p1[8];
#pragma unroll
      for (int r = 0; r < 8; ++r) {
        int rowg = qi + r + 8 * half;
        float v0 = ((kb + ln) <= rowg) ? s[i][0][r] * 0.125f : -INFINITY;
        float v1 = (act1 && (kb + 16 + ln) <= rowg) ? s[i][1][r] * 0.125f
                                                    : -INFINITY;
        float mx = fmaxf(v0, v1);
        mx = fmaxf(mx, __shfl_xor(mx, 1, 32));
        mx = fmaxf(mx, __shfl_xor(mx, 2, 32));
        mx = fmaxf(mx, __shfl_xor(mx, 4, 32));
        mx = fmaxf(mx, __shfl_xor(mx, 8, 32));
        float mn = fmaxf(m_i[i][r], mx);
        float sc = __expf(m_i[i][r] - mn);   // exp(-inf)=0 on first block
        m_i[i][r] = mn;
        p0[r] = __expf(v0 - mn);
        p1[r] = __expf(v1 - mn);
        float rs = p0[r] + p1[r];
        rs += __shfl_xor(rs, 1, 32);
        rs += __shfl_xor(rs, 2, 32);
        rs += __shfl_xor(rs, 4, 32);
        rs += __shfl_xor(rs, 8, 32);
        l_i[i][r] = l_i[i][r] * sc + rs;
#pragma unroll
        for (int j = 0; j < 4; ++j) acc[i][j][r] *= sc;
      }

      // D-layout -> A-layout relayout of P (16x32) via per-wave LDS.
      // Same-wave DS ops are in-order (ISA 7.3): compiler fence suffices.
#pragma unroll
      for (int r = 0; r < 8; ++r) {
        int row = r + 8 * half;
        lds[wv][i][row * 32 + ln] = (_Float16)p0[r];
        lds[wv][i][row * 32 + 16 + ln] = (_Float16)p1[r];
      }
      asm volatile("" ::: "memory");
      v8h plo = *(const v8h*)&lds[wv][i][ln * 32 + half * 16];
      v8h phi = *(const v8h*)&lds[wv][i][ln * 32 + half * 16 + 8];
      v16h ap = combine8(plo, phi);
      asm volatile("" ::: "memory");

#pragma unroll
      for (int j = 0; j < 4; ++j) acc[i][j] = wmma_f16(ap, vf[j], acc[i][j]);
    }
  }

#pragma unroll
  for (int i = 0; i < 2; ++i) {
#pragma unroll
    for (int r = 0; r < 8; ++r) {
      float inv = 1.0f / l_i[i][r];
      int rowg = q0 + i * 16 + r + 8 * half;
      size_t base = ((size_t)b_ * TT + rowg) * CC + head * HDD;
#pragma unroll
      for (int j = 0; j < 4; ++j)
        Yh[base + j * 16 + ln] = (_Float16)(acc[i][j][r] * inv);
    }
  }
}

// ---------------- output projection GEMM ----------------
// Yh: (M=8192, K=768) f16 ; Wt: (N=768, K=768) f16 ; out f32 + bias.
// Same LDS-staged structure as the QKV GEMM.
__global__ __launch_bounds__(256) void k_proj_gemm(
    const _Float16* __restrict__ Yh, const _Float16* __restrict__ Wt,
    const float* __restrict__ bias, float* __restrict__ out) {
  __shared__ __align__(64) _Float16 ldsB[2][128][LDSPITCH];
  const int tid = threadIdx.x;
  const int l = tid & 31;
  const int wv = tid >> 5;
  const int ln = l & 15, half = l >> 4;
  const int m0 = blockIdx.x * 128 + (wv & 3) * 32;
  const int n0b = blockIdx.y * 128;
  const int nf = (wv >> 2) * 64;

  const _Float16* Ap0 = Yh + (size_t)(m0 + ln) * CC;
  const _Float16* Ap1 = Yh + (size_t)(m0 + 16 + ln) * CC;
  const v8f zero = {0.f, 0.f, 0.f, 0.f, 0.f, 0.f, 0.f, 0.f};
  v8f acc[2][4] = {{zero, zero, zero, zero}, {zero, zero, zero, zero}};

  auto stage = [&](int buf, int k0) {
#pragma unroll
    for (int s = 0; s < 2; ++s) {
      int i = tid + 256 * s;
      int nrow = i >> 2, seg = i & 3;
      const _Float16* gp = Wt + (size_t)(n0b + nrow) * CC + k0 + seg * 8;
      unsigned la = (unsigned)(size_t)&ldsB[buf][nrow][seg * 8];
      async_copy_b128(gp, la);
    }
  };

  stage(0, 0);
  for (int k0 = 0; k0 < CC; k0 += 32) {
    int buf = (k0 >> 5) & 1;
    wait_async0();
    __syncthreads();
    if (k0 + 32 < CC) stage(buf ^ 1, k0 + 32);

    v16h a0 = load_afrag(Ap0, k0, half);
    v16h a1 = load_afrag(Ap1, k0, half);
#pragma unroll
    for (int j = 0; j < 4; ++j) {
      v16h b = *(const v16h*)&ldsB[buf][nf + j * 16 + ln][half * 16];
      acc[0][j] = wmma_f16(a0, b, acc[0][j]);
      acc[1][j] = wmma_f16(a1, b, acc[1][j]);
    }
    __syncthreads();
  }

#pragma unroll
  for (int j = 0; j < 4; ++j) {
    int n = n0b + nf + j * 16 + ln;
    float bv = bias[n];
#pragma unroll
    for (int i = 0; i < 2; ++i) {
#pragma unroll
      for (int r = 0; r < 8; ++r) {
        int row = m0 + i * 16 + r + 8 * half;
        out[(size_t)row * CC + n] = acc[i][j][r] + bv;
      }
    }
  }
}

// ---------------- launch ----------------
extern "C" void kernel_launch(void* const* d_in, const int* in_sizes, int n_in,
                              void* d_out, int out_size, void* d_ws,
                              size_t ws_size, hipStream_t stream) {
  const float* x = (const float*)d_in[0];
  const float* W_attn = (const float*)d_in[1];
  const float* b_attn = (const float*)d_in[2];
  const float* W_proj = (const float*)d_in[3];
  const float* b_proj = (const float*)d_in[4];
  float* out = (float*)d_out;

  char* ws = (char*)d_ws;
  _Float16* xh = (_Float16*)ws;       ws += (size_t)MM * CC * 2;
  _Float16* Wqkv_t = (_Float16*)ws;   ws += (size_t)N3 * CC * 2;
  _Float16* Wproj_t = (_Float16*)ws;  ws += (size_t)CC * CC * 2;
  _Float16* Qd = (_Float16*)ws;       ws += (size_t)BB * NHH * TT * HDD * 2;
  _Float16* Kd = (_Float16*)ws;       ws += (size_t)BB * NHH * TT * HDD * 2;
  _Float16* Vtd = (_Float16*)ws;      ws += (size_t)BB * NHH * HDD * TT * 2;
  _Float16* Yh = (_Float16*)ws;       ws += (size_t)MM * CC * 2;

  int nx = MM * CC;
  k_f32_to_f16<<<(nx + 255) / 256, 256, 0, stream>>>(x, xh, nx);
  int nw1 = CC * N3;
  k_transpose_f16<<<(nw1 + 255) / 256, 256, 0, stream>>>(W_attn, Wqkv_t, CC, N3);
  int nw2 = CC * CC;
  k_transpose_f16<<<(nw2 + 255) / 256, 256, 0, stream>>>(W_proj, Wproj_t, CC, CC);

  k_qkv_gemm<<<dim3(MM / 128, N3 / 128), 256, 0, stream>>>(xh, Wqkv_t, b_attn,
                                                           Qd, Kd, Vtd);

  int tasks = BB * NHH * (TT / 32);   // 3072 waves
  k_attn<<<tasks / 4, 128, 0, stream>>>(Qd, Kd, Vtd, Yh);

  k_proj_gemm<<<dim3(MM / 128, CC / 128), 256, 0, stream>>>(Yh, Wproj_t,
                                                            b_proj, out);
}
